// EloisHead_14293651161278
// MI455X (gfx1250) — matrix operand order repaired and checked by reading, not compiled, not run on hardware
//
#include <hip/hip_runtime.h>
#include <hip/hip_bf16.h>
#include <stdint.h>

#define B_TOTAL 16384
#define E       1024
#define H       128
#define TB      64              // batches per workgroup
#define MROWS   128             // 2*TB rows of X2 = [2B, E]
#define KC      32              // K chunk per WMMA step
#define NK      (E / KC)        // 32 chunks
#define SSTR    40              // staging row stride in ushort (80B, conflict-free)
#define YSTR    132             // Y row stride in floats (padded)

typedef __attribute__((ext_vector_type(16))) __bf16 v16bf;
typedef __attribute__((ext_vector_type(2)))  __bf16 v2bf;
typedef __attribute__((ext_vector_type(2)))  float  v2f;
typedef __attribute__((ext_vector_type(8)))  float  v8f;

union Frag { v16bf v; uint4 q[2]; };

#if defined(__has_builtin)
#if __has_builtin(__builtin_amdgcn_sched_group_barrier)
#define SCHED_GROUP(mask, size, id) __builtin_amdgcn_sched_group_barrier((mask), (size), (id))
#endif
#endif
#ifndef SCHED_GROUP
#define SCHED_GROUP(mask, size, id)
#endif

// pack two fp32 -> one dword of 2x bf16 (RNE); vector convert so the backend
// can select a single packed v_cvt_pk_bf16_f32.
__device__ __forceinline__ uint32_t pack2bf(float a, float b) {
  v2f f;
  f.x = a; f.y = b;
  union { v2bf h; uint32_t u; } v;
  v.h = __builtin_convertvector(f, v2bf);
  return v.u;
}

__global__ __launch_bounds__(256, 2)
void elois_head_fused(const float* __restrict__ x,
                      const float* __restrict__ Wm,
                      const float* __restrict__ cov,
                      const float* __restrict__ bias,
                      float* __restrict__ out)
{
  // LDS union: phase A staging (2x (128x40) ushort for X and W = 40 KB)
  //            phase B Y matrix (128 x 132 fp32 = 66 KB)
  __shared__ __align__(16) unsigned char smem[67584];
  unsigned short* sX = (unsigned short*)smem;                   // [2][128][SSTR]
  unsigned short* sW = (unsigned short*)(smem + 20480);         // [2][128][SSTR]
  float*          Yf = (float*)smem;                            // [128][YSTR]

  const int tid  = threadIdx.x;
  const int wv   = tid >> 5;          // wave 0..7
  const int lane = tid & 31;
  const int b0   = blockIdx.x * TB;   // first batch of this tile

  // ---------------- Phase A: Y[0:128][0:128] = bf16( X2 tile ) @ bf16( W )^T ---------------
  const int srow = tid >> 1;                 // 0..127
  const int scol = (tid & 1) * 16;           // 0 or 16 within K chunk

  const float* xbase = x  + (long)(b0 * 2 + srow) * E + scol;   // X2 row-major
  const float* wbase = Wm + (long)srow * E + scol;              // W  row-major [H,E]

  float4 xr[4], wr[4];
  auto loadRegs = [&](int i) {
    const float* xp = xbase + i * KC;
    const float* wp = wbase + i * KC;
#pragma unroll
    for (int q = 0; q < 4; q++) {
      xr[q] = ((const float4*)xp)[q];
      wr[q] = ((const float4*)wp)[q];
    }
  };
  auto writeLDS = [&](int buf) {
    unsigned short* dx = sX + buf * (MROWS * SSTR) + srow * SSTR + scol;
    unsigned short* dw = sW + buf * (MROWS * SSTR) + srow * SSTR + scol;
    uint32_t px[8], pw[8];
#pragma unroll
    for (int q = 0; q < 4; q++) {
      px[2*q+0] = pack2bf(xr[q].x, xr[q].y);
      px[2*q+1] = pack2bf(xr[q].z, xr[q].w);
      pw[2*q+0] = pack2bf(wr[q].x, wr[q].y);
      pw[2*q+1] = pack2bf(wr[q].z, wr[q].w);
    }
    ((uint4*)dx)[0] = make_uint4(px[0], px[1], px[2], px[3]);
    ((uint4*)dx)[1] = make_uint4(px[4], px[5], px[6], px[7]);
    ((uint4*)dw)[0] = make_uint4(pw[0], pw[1], pw[2], pw[3]);
    ((uint4*)dw)[1] = make_uint4(pw[4], pw[5], pw[6], pw[7]);
  };

  // WMMA fragment addressing (16-bit A 16x32 layout per CDNA5 ISA 7.12.2)
  const int arow = 16 * wv + (lane & 15);     // wave owns output rows [16w,16w+16)
  const int k0   = (lane >> 4) << 3;          // A: lanes 0-15 K=0.., lanes 16-31 K=8..
  const int kb   = (lane >> 4) << 3;          // B: same K order as A (lanes 0-15 K=0..7,16..23; lanes 16-31 K=8..15,24..31)
  const int brow = lane & 15;

  v8f acc[8] = {};                            // 16 rows x 128 cols in 8 tiles

  auto computeChunk = [&](int buf) {
    const unsigned short* pA = sX + buf * (MROWS * SSTR) + arow * SSTR;
    Frag a;
    a.q[0] = *(const uint4*)(pA + k0);        // K = k0 .. k0+7
    a.q[1] = *(const uint4*)(pA + 16 + k0);   // K = 16+k0 .. 16+k0+7
    const unsigned short* pWb = sW + buf * (MROWS * SSTR);
    Frag b[8];
#pragma unroll
    for (int t = 0; t < 8; t++) {
      const unsigned short* pB = pWb + (16 * t + brow) * SSTR + kb;
      b[t].q[0] = ((const uint4*)pB)[0];
      b[t].q[1] = ((const uint4*)pB)[2];
    }
#pragma unroll
    for (int t = 0; t < 8; t++) {
      acc[t] = __builtin_amdgcn_wmma_f32_16x16x32_bf16(
          /*neg_a=*/false, a.v, /*neg_b=*/false, b[t].v,
          /*c_mod=*/(short)0, acc[t], /*reuse_a=*/false, /*reuse_b=*/false);
    }
    // Scheduling pin: keep a 2-tile DS-read lookahead ahead of each WMMA so
    // waits become dscnt<=4 instead of dscnt==0 (LDS latency hides under the
    // matrix pipe).  masks: 0x100 = DS read, 0x008 = matrix (WMMA).
    SCHED_GROUP(0x100, 6, 0);                 // A(2) + B0(2) + B1(2)
#pragma unroll
    for (int t = 0; t < 6; t++) {
      SCHED_GROUP(0x008, 1, 0);               // WMMA t
      SCHED_GROUP(0x100, 2, 0);               // B(t+2)
    }
    SCHED_GROUP(0x008, 2, 0);                 // WMMA 6,7
  };

  // software-pipelined, double-buffered K loop (one barrier per chunk)
  loadRegs(0);
  writeLDS(0);
  loadRegs(1);
  __syncthreads();
#pragma unroll 1
  for (int i = 0; i < NK; i++) {
    computeChunk(i & 1);
    if (i + 1 < NK) writeLDS((i + 1) & 1);
    if (i + 2 < NK) loadRegs(i + 2);
    __syncthreads();
  }

  // spill C fragments to LDS Y (row 2i = values of batch i, row 2i+1 = mask)
  {
    const int rbase = 16 * wv + ((lane >> 4) << 3);
    const int cbase = lane & 15;
#pragma unroll
    for (int t = 0; t < 8; t++)
#pragma unroll
      for (int v = 0; v < 8; v++)
        Yf[(rbase + v) * YSTR + 16 * t + cbase] = acc[t][v];
  }
  __syncthreads();

  // ---------------- Phase B: out[b,k] = sum_h values[b,h]*softmax_k(mask[b,h]*cov[h,k]) + bias[k]
  // wave handles 8 batches; each lane owns k = lane + 32*j, j=0..3
  float bsv[4];
#pragma unroll
  for (int j = 0; j < 4; j++) bsv[j] = bias[lane + 32 * j];

  float o[8][4];
#pragma unroll
  for (int ib = 0; ib < 8; ib++)
#pragma unroll
    for (int j = 0; j < 4; j++) o[ib][j] = 0.f;

#pragma unroll 1
  for (int h = 0; h < H; h++) {
    float cv[4];
#pragma unroll
    for (int j = 0; j < 4; j++) cv[j] = cov[h * H + lane + 32 * j];  // L2-resident, coalesced
#pragma unroll
    for (int ib = 0; ib < 8; ib++) {
      const int li  = 8 * wv + ib;
      float vals = Yf[(2 * li) * YSTR + h];       // LDS broadcast
      float msk  = Yf[(2 * li + 1) * YSTR + h];   // LDS broadcast
      float e[4], psum = 0.f;
#pragma unroll
      for (int j = 0; j < 4; j++) { e[j] = __expf(msk * cv[j]); psum += e[j]; }
#pragma unroll
      for (int off = 16; off >= 1; off >>= 1) psum += __shfl_xor(psum, off, 32);
      float s = __fdividef(vals, psum);           // values / Z  (softmax normalize folded in)
#pragma unroll
      for (int j = 0; j < 4; j++) o[ib][j] = fmaf(s, e[j], o[ib][j]);
    }
  }

  // coalesced store (+bias)
#pragma unroll
  for (int ib = 0; ib < 8; ib++) {
    long b = b0 + 8 * wv + ib;
#pragma unroll
    for (int j = 0; j < 4; j++)
      out[b * H + lane + 32 * j] = o[ib][j] + bsv[j];
  }
}

extern "C" void kernel_launch(void* const* d_in, const int* in_sizes, int n_in,
                              void* d_out, int out_size, void* d_ws, size_t ws_size,
                              hipStream_t stream) {
  const float* x    = (const float*)d_in[0];
  const float* Wm   = (const float*)d_in[1];
  const float* cov  = (const float*)d_in[2];
  const float* bias = (const float*)d_in[3];
  float* out = (float*)d_out;
  dim3 grid(B_TOTAL / TB);   // 256 workgroups
  dim3 block(256);           // 8 wave32 waves
  elois_head_fused<<<grid, block, 0, stream>>>(x, Wm, cov, bias, out);
}
